// QuantumFeedForwardLayer_65481071407974
// MI455X (gfx1250) — matrix operand order repaired and checked by reading, not compiled, not run on hardware
//
#include <hip/hip_runtime.h>

// Fused QuantumFeedForward layer for MI455X (gfx1250, wave32, WMMA + TDM).
// B=16, S=4096, D=256  ->  M = 65536 rows, two 65536x256x256 GEMMs fused.
//  - x tile staged LDS via TENSOR_LOAD_TO_LDS (one DMA per workgroup)
//  - weights pre-packed once into f16 WMMA B-fragment layout in d_ws
//  - GEMM inner loop: ds_load_b128 + global_load_b128 + v_wmma_f32_16x16x32_f16

#if __has_builtin(__builtin_amdgcn_tensor_load_to_lds)
#define QFFN_HAVE_TDM 1
#else
#define QFFN_HAVE_TDM 0
#endif

namespace {

constexpr int kD     = 256;        // embed dim = ffn dim = K = N
constexpr int kBM    = 64;         // rows per workgroup
constexpr int kMT    = kBM / 16;   // 4 M-tiles of 16
constexpr int kKT    = kD / 32;    // 8 K-steps of 32 (f16 WMMA K)
constexpr int kNT    = kD / 16;    // 16 N-tiles
constexpr int kWaves = 8;          // 256 threads / wave32

// packed-weight buffer: unsigned pw[2][kNT][kKT][32][8]
constexpr int    kFragDw      = 32 * 8;                    // dwords per fragment
constexpr int    kWDwords     = kNT * kKT * kFragDw;       // per weight matrix
constexpr size_t kPackedBytes = (size_t)2 * kWDwords * 4;  // 256 KB

typedef _Float16 v16h __attribute__((ext_vector_type(16)));
typedef float    v8f  __attribute__((ext_vector_type(8)));
typedef unsigned v4u  __attribute__((ext_vector_type(4)));
typedef int      v4i  __attribute__((ext_vector_type(4)));
typedef int      v8i  __attribute__((ext_vector_type(8)));

union Frag {
  unsigned u[8];
  uint4    q[2];
  v16h     v;
};

__device__ inline unsigned packh2(float a, float b) {
  union { unsigned u; _Float16 h[2]; } r;
  r.h[0] = (_Float16)a;
  r.h[1] = (_Float16)b;
  return r.u;
}

} // namespace

//============================================================================
// One-shot pre-pack: W (f32 row-major [K][N]) -> f16 B-fragment layout.
// B-fragment (32x16 f16): lane L holds col N=L&15; dword v holds K-pair
// {16*(L>>4) + 2v, +1}.
//============================================================================
__global__ __launch_bounds__(256)
void pack_weights_kernel(const float* __restrict__ W1,
                         const float* __restrict__ W2,
                         unsigned* __restrict__ pw)
{
  const int idx  = blockIdx.x * 256 + threadIdx.x;   // 0 .. 2*kWDwords-1
  const int v    = idx & 7;
  const int lane = (idx >> 3) & 31;
  const int kt   = (idx >> 8) & (kKT - 1);
  const int nt   = (idx >> 11) & (kNT - 1);
  const int w    = idx >> 15;
  const int n    = nt * 16 + (lane & 15);
  const int k    = kt * 32 + 16 * (lane >> 4) + 2 * v;
  const float* W = w ? W2 : W1;
  pw[idx] = packh2(W[(size_t)k * kD + n], W[(size_t)(k + 1) * kD + n]);
}

//============================================================================
// Fused main kernel.  PACKED=true loads B-fragments from pw (b128 loads);
// PACKED=false builds them inline from f32 weights (fallback if ws too small).
//============================================================================
template <bool PACKED>
__global__ __launch_bounds__(256)
void qffn_fused_kernel(const float* __restrict__ x,
                       const float* __restrict__ W1,
                       const float* __restrict__ b1,
                       const float* __restrict__ W2,
                       const float* __restrict__ b2,
                       const float* __restrict__ qwp,
                       const unsigned* __restrict__ pw,
                       float* __restrict__ out)
{
  // 64KB: raw x tile, staged by the Tensor Data Mover
  __shared__ float sX[kBM][kD];
  // 32KB: A-fragments in WMMA layout (holds x-frags, then reused for mixed)
  __shared__ unsigned sA[kMT][kKT][32][8];
  // 32KB: quantum branch values q[m][n] (f16)
  __shared__ _Float16 sQ[kBM][kD];

  const int    tid  = threadIdx.x;
  const int    wv   = tid >> 5;
  const int    lane = tid & 31;
  const size_t row0 = (size_t)blockIdx.x * kBM;

  //--------------------------------------------------------------------------
  // Phase -1: TDM stage of the x tile (64 x 256 f32, contiguous 64KB).
  // One wave issues the DMA (EXEC-independent), waits TENSORcnt, then the
  // workgroup barrier publishes sX to all waves.
  //--------------------------------------------------------------------------
#if QFFN_HAVE_TDM
  if (wv == 0) {
    const unsigned           lds = (unsigned)(uintptr_t)&sX[0][0];
    const unsigned long long ga  = (unsigned long long)(uintptr_t)(x + row0 * kD);
    // D# group0: count=1 | lds_addr | global_addr[56:0] | type=2
    v4u g0 = { 1u,
               lds,
               (unsigned)(ga & 0xffffffffu),
               (unsigned)((ga >> 32) & 0x01ffffffu) | (2u << 30) };
    // D# group1: data_size=4B; tensor_dim0=256; tensor_dim1=64;
    //            tile_dim0=256; tile_dim1=64; tensor_dim0_stride=256
    v8i g1 = { (int)(2u << 16),        // data_size=2 (4 bytes) at bits[17:16]
               (int)(256u << 16),      // tensor_dim0[15:0] at bits[79:64... d1[31:16]]
               (int)(64u  << 16),      // tensor_dim1[15:0] at d2[31:16]
               (int)(256u << 16),      // tile_dim0 at d3[31:16]
               64,                     // tile_dim1 at d4[15:0]
               256,                    // tensor_dim0_stride low at d5[31:0]
               0, 0 };
    v4i gz = { 0, 0, 0, 0 };           // groups 2/3: dims >=2 unused
#if defined(__clang_major__) && (__clang_major__ >= 23)
    v8i gz8 = { 0, 0, 0, 0, 0, 0, 0, 0 };
    __builtin_amdgcn_tensor_load_to_lds(g0, g1, gz, gz, gz8, 0);
#else
    __builtin_amdgcn_tensor_load_to_lds(g0, g1, gz, gz, 0);
#endif
    __builtin_amdgcn_s_wait_tensorcnt(0);
  }
  __syncthreads();
#endif

  //--------------------------------------------------------------------------
  // Phase 0a: quantum branch  q[m][n] = prod_{i<=n} cos(x[m][i])
  //--------------------------------------------------------------------------
  for (int r = wv; r < kBM; r += kWaves) {
#if QFFN_HAVE_TDM
    const float4* xr = (const float4*)&sX[r][lane * 8];
#else
    const float4* xr = (const float4*)(x + (row0 + r) * kD + lane * 8);
#endif
    float4 a0 = xr[0];
    float4 a1 = xr[1];
    float c[8] = { __cosf(a0.x), __cosf(a0.y), __cosf(a0.z), __cosf(a0.w),
                   __cosf(a1.x), __cosf(a1.y), __cosf(a1.z), __cosf(a1.w) };
    float p[8];
    p[0] = c[0];
#pragma unroll
    for (int j = 1; j < 8; ++j) p[j] = p[j - 1] * c[j];

    float s = p[7];                       // wave32 inclusive scan
#pragma unroll
    for (int off = 1; off < 32; off <<= 1) {
      float t = __shfl_up(s, off, 32);
      if (lane >= off) s *= t;
    }
    float ex = __shfl_up(s, 1, 32);       // -> exclusive
    if (lane == 0) ex = 1.0f;

    union { uint4 q; _Float16 h[8]; } o;
#pragma unroll
    for (int j = 0; j < 8; ++j) o.h[j] = (_Float16)(ex * p[j]);
    *(uint4*)&sQ[r][lane * 8] = o.q;
  }

  //--------------------------------------------------------------------------
  // Phase 0b: pack x into f16 WMMA A-fragments in contiguous k order (k = 16*hi + 2v, +1).
  //--------------------------------------------------------------------------
#pragma unroll
  for (int it = 0; it < (kMT * kKT * 32 * 8) / 256; ++it) {
    int idx = tid + it * 256;
    int v   = idx & 7;
    int ln  = (idx >> 3) & 31;
    int kt  = (idx >> 8) & (kKT - 1);
    int mt  = idx >> 11;
    int m   = mt * 16 + (ln & 15);
    int hi  = ln >> 4;
    int kb  = kt * 32 + 16 * hi + 2 * v;
#if QFFN_HAVE_TDM
    float2 xv = *(const float2*)&sX[m][kb];
#else
    float2 xv = *(const float2*)(x + (row0 + m) * kD + kb);
#endif
    sA[mt][kt][ln][v] = packh2(xv.x, xv.y);
  }
  __syncthreads();

  //--------------------------------------------------------------------------
  // Phase 1: classical = relu(x @ W1 + b1).  Wave w owns cols [32w, 32w+32).
  //--------------------------------------------------------------------------
  const int n0 = wv * 32;
  const v8f vzero = {0.f, 0.f, 0.f, 0.f, 0.f, 0.f, 0.f, 0.f};
  const unsigned* pw1 = pw;
  const unsigned* pw2 = pw + kWDwords;

  v8f acc[kMT][2];
#pragma unroll
  for (int mt = 0; mt < kMT; ++mt) { acc[mt][0] = vzero; acc[mt][1] = vzero; }

  for (int kt = 0; kt < kKT; ++kt) {
    if (kt + 1 < kKT) {
      if (PACKED)
        __builtin_prefetch(pw1 + ((size_t)((2 * wv) * kKT + kt + 1) * 32 + lane) * 8, 0, 3);
      else
        __builtin_prefetch(&W1[(size_t)((kt + 1) * 32 + 16 * (lane >> 4)) * kD +
                               n0 + (lane & 15)], 0, 3);
    }
    Frag a[kMT];
#pragma unroll
    for (int mt = 0; mt < kMT; ++mt) {
      a[mt].q[0] = *(const uint4*)&sA[mt][kt][lane][0];
      a[mt].q[1] = *(const uint4*)&sA[mt][kt][lane][4];
    }
#pragma unroll
    for (int j = 0; j < 2; ++j) {
      const int nt = 2 * wv + j;
      Frag bf;
      if constexpr (PACKED) {
        const uint4* src = (const uint4*)(pw1 + ((size_t)(nt * kKT + kt) * 32 + lane) * 8);
        bf.q[0] = src[0];
        bf.q[1] = src[1];
      } else {
        const int n     = nt * 16 + (lane & 15);
        const int kbase = kt * 32 + 16 * (lane >> 4);
#pragma unroll
        for (int v = 0; v < 8; ++v) {
          const int k = kbase + 2 * v;
          bf.u[v] = packh2(W1[(size_t)k * kD + n], W1[(size_t)(k + 1) * kD + n]);
        }
      }
#pragma unroll
      for (int mt = 0; mt < kMT; ++mt)
        acc[mt][j] = __builtin_amdgcn_wmma_f32_16x16x32_f16(
            false, a[mt].v, false, bf.v, (short)0, acc[mt][j], false, false);
    }
  }
  __syncthreads();   // all waves done reading x-fragments from sA

  //--------------------------------------------------------------------------
  // Phase 1 epilogue: bias + relu, mix with quantum branch, scatter mixed
  // value into A-fragment layout (reusing sA) for GEMM2.
  // C/D layout: VGPR e -> M = e + 8*(lane>>4), N = lane&15.
  //--------------------------------------------------------------------------
  const float wq = 1.0f / (1.0f + __expf(-qwp[0]));
  const float wc = 1.0f - wq;
#pragma unroll
  for (int j = 0; j < 2; ++j) {
    const int   n    = n0 + j * 16 + (lane & 15);
    const float bias = b1[n];
    const int kt2 = n >> 5;             // inverse A-fragment map for K = n
    const int ko  = n & 31;
    const int pp  = ko >> 1;
    const int h   = ko & 1;
    const int hi2 = pp >> 3;
    const int v2  = pp & 7;
#pragma unroll
    for (int mt = 0; mt < kMT; ++mt) {
#pragma unroll
      for (int e = 0; e < 8; ++e) {
        const int m  = mt * 16 + e + 8 * (lane >> 4);
        float cls = acc[mt][j][e] + bias;
        cls = cls > 0.0f ? cls : 0.0f;
        const float qv = (float)sQ[m][n];
        const float mx = wq * qv + wc * cls;
        _Float16* dst = (_Float16*)&sA[m >> 4][kt2][(m & 15) + 16 * hi2][v2];
        dst[h] = (_Float16)mx;
      }
    }
  }
  __syncthreads();   // mixed fragments ready

  //--------------------------------------------------------------------------
  // Phase 2: out = mixed @ W2 + b2
  //--------------------------------------------------------------------------
  v8f acc2[kMT][2];
#pragma unroll
  for (int mt = 0; mt < kMT; ++mt) { acc2[mt][0] = vzero; acc2[mt][1] = vzero; }

  for (int kt = 0; kt < kKT; ++kt) {
    if (kt + 1 < kKT) {
      if (PACKED)
        __builtin_prefetch(pw2 + ((size_t)((2 * wv) * kKT + kt + 1) * 32 + lane) * 8, 0, 3);
      else
        __builtin_prefetch(&W2[(size_t)((kt + 1) * 32 + 16 * (lane >> 4)) * kD +
                               n0 + (lane & 15)], 0, 3);
    }
    Frag a[kMT];
#pragma unroll
    for (int mt = 0; mt < kMT; ++mt) {
      a[mt].q[0] = *(const uint4*)&sA[mt][kt][lane][0];
      a[mt].q[1] = *(const uint4*)&sA[mt][kt][lane][4];
    }
#pragma unroll
    for (int j = 0; j < 2; ++j) {
      const int nt = 2 * wv + j;
      Frag bf;
      if constexpr (PACKED) {
        const uint4* src = (const uint4*)(pw2 + ((size_t)(nt * kKT + kt) * 32 + lane) * 8);
        bf.q[0] = src[0];
        bf.q[1] = src[1];
      } else {
        const int n     = nt * 16 + (lane & 15);
        const int kbase = kt * 32 + 16 * (lane >> 4);
#pragma unroll
        for (int v = 0; v < 8; ++v) {
          const int k = kbase + 2 * v;
          bf.u[v] = packh2(W2[(size_t)k * kD + n], W2[(size_t)(k + 1) * kD + n]);
        }
      }
#pragma unroll
      for (int mt = 0; mt < kMT; ++mt)
        acc2[mt][j] = __builtin_amdgcn_wmma_f32_16x16x32_f16(
            false, a[mt].v, false, bf.v, (short)0, acc2[mt][j], false, false);
    }
  }

  // final epilogue: add b2, store f32
#pragma unroll
  for (int j = 0; j < 2; ++j) {
    const int   n    = n0 + j * 16 + (lane & 15);
    const float bias = b2[n];
#pragma unroll
    for (int mt = 0; mt < kMT; ++mt) {
#pragma unroll
      for (int e = 0; e < 8; ++e) {
        const int m = mt * 16 + e + 8 * (lane >> 4);
        out[(row0 + m) * kD + n] = acc2[mt][j][e] + bias;
      }
    }
  }
}

extern "C" void kernel_launch(void* const* d_in, const int* in_sizes, int n_in,
                              void* d_out, int out_size, void* d_ws, size_t ws_size,
                              hipStream_t stream) {
  (void)in_sizes; (void)n_in; (void)out_size;
  const float* x  = (const float*)d_in[0];
  const float* W1 = (const float*)d_in[1];
  const float* b1 = (const float*)d_in[2];
  const float* W2 = (const float*)d_in[3];
  const float* b2 = (const float*)d_in[4];
  // d_in[5] = param (RZ phases, provably no effect on output)
  const float* qw = (const float*)d_in[6];
  float* out = (float*)d_out;

  const int M = 16 * 4096;                 // B * S
  dim3 grid(M / kBM);                      // 1024 workgroups
  dim3 block(256);                         // 8 wave32 waves

  if (ws_size >= kPackedBytes && d_ws != nullptr) {
    unsigned* pw = (unsigned*)d_ws;
    // one-shot weight pre-pack into f16 B-fragment layout (L2-resident)
    pack_weights_kernel<<<dim3(2 * kWDwords / 256), block, 0, stream>>>(W1, W2, pw);
    qffn_fused_kernel<true><<<grid, block, 0, stream>>>(x, W1, b1, W2, b2, qw, pw, out);
  } else {
    qffn_fused_kernel<false><<<grid, block, 0, stream>>>(x, W1, b1, W2, b2, qw, nullptr, out);
  }
}
